// DynamicPerPixelConv2d_1529008358152
// MI455X (gfx1250) — matrix operand order repaired and checked by reading, not compiled, not run on hardware
//
#include <hip/hip_runtime.h>
#include <math.h>

typedef __attribute__((ext_vector_type(16))) _Float16 v16h;
typedef __attribute__((ext_vector_type(8)))  _Float16 v8h;
typedef __attribute__((ext_vector_type(8)))  float    v8f;

namespace {
constexpr int CIN   = 64;
constexpr int OUTC  = 64;
constexpr int CPG   = 16;                // CIN / PG
constexpr int KS    = 3;
constexpr int HH    = 48;
constexpr int WW    = 48;
constexpr int LPIX  = HH * WW;           // 2304
constexpr int CKK   = CIN * KS * KS;     // 576
constexpr int WSZ   = 16 * CKK;          // 9216  (BOT * CKK)
constexpr int PSZ   = OUTC * 16;         // 1024
constexpr int GRP   = 2576;              // PRED_CH / PG (multiple of 16)
constexpr int TPX   = 16;                // pixels per tile
constexpr int TROW  = WW / TPX;          // 3
constexpr int NTILE = LPIX / TPX;        // 144
constexpr int NB    = 2;

// ---- LDS layout (bank-conflict padded; 64 dword banks) ----
constexpr int OFF_XPAD = 0;              // float [64][3][18]          (13824 B)
constexpr int OFF_XCOL = 13824;          // f16   [64][16]             ( 2048 B)
constexpr int OFF_PROJ = 15872;          // f16, px-stride 1048 halves (33536 B)
constexpr int PROJ_PXH = 1048;           //   524 dw, %64 == 12 -> distinct banks
constexpr int OFF_DYNB = 49408;          // f32, px-stride 68 floats   ( 4352 B)
constexpr int DYNB_PXF = 68;             //   68 dw, %64 == 4
constexpr int OFF_PW   = 53760;          // f16 [px][j][d]             (12544 B)
constexpr int PW_JH    = 24;             //   j-stride 12 dw
constexpr int PW_PXH   = 392;            //   px-stride 196 dw, %64 == 4
constexpr int OFF_DOT  = 66304;          // float [16][64][9]          (36864 B)
constexpr int OFF_SQ   = 103168;         // float [16][64][9]          (36864 B)
constexpr int SMEM_BYTES = 140032;       // ~136.8 KB < 160 KB -> 2 WGs per WGP
}

// A-fragment for v_wmma_f32_16x16x32_f16 from 16 consecutive fp32 rows of Wp.
// Real K = 16 (halves 0..7 per lane-half), K=16..31 zero-padded (halves 8..15).
__device__ __forceinline__ v16h wp_afrag(const float* __restrict__ Wp, int r0, int lane) {
    const int row = r0 + (lane & 15);
    const float* wr = Wp + (size_t)row * CPG + ((lane & 16) ? 8 : 0);
    v16h a = {};
#pragma unroll
    for (int h = 0; h < 8; ++h) a[h] = (_Float16)wr[h];   // 2x global_load_b128 + cvt_pk
    return a;
}

__global__ __launch_bounds__(128, 2)
void dynconv_fused(const float* __restrict__ X, const float* __restrict__ Wp,
                   const float* __restrict__ bp, float* __restrict__ Out)
{
    extern __shared__ __align__(16) char smem[];
    float    (*s_xpad)[3][18] = reinterpret_cast<float   (*)[3][18]>(smem + OFF_XPAD);
    _Float16 (*s_xcol)[16]    = reinterpret_cast<_Float16(*)[16]   >(smem + OFF_XCOL);
    _Float16* s_proj          = reinterpret_cast<_Float16*>(smem + OFF_PROJ);
    float*    s_dynb          = reinterpret_cast<float*   >(smem + OFF_DYNB);
    _Float16* s_pw            = reinterpret_cast<_Float16*>(smem + OFF_PW);
    float    (*s_dot)[64][9]  = reinterpret_cast<float   (*)[64][9]>(smem + OFF_DOT);
    float    (*s_sq)[64][9]   = reinterpret_cast<float   (*)[64][9]>(smem + OFF_SQ);

    const int tid  = threadIdx.x;
    const int lane = tid & 31;           // wave32
    const int wave = tid >> 5;           // 4 waves / block
    const int bi = blockIdx.x / NTILE;
    const int t  = blockIdx.x % NTILE;
    const int y  = t / TROW;
    const int x0 = (t % TROW) * TPX;

    const float* xb = X + (size_t)bi * CIN * LPIX;

    // ---- Phase 0: stage padded 3x18 neighborhood + zero accumulators
    for (int i = tid; i < CIN * 3 * 18; i += blockDim.x) {
        const int c  = i / 54;
        const int rm = i % 54;
        const int ry = rm / 18;
        const int rx = rm % 18;
        const int gy = y - 1 + ry;
        const int gx = x0 - 1 + rx;
        float v = 0.f;
        if ((unsigned)gy < (unsigned)HH && (unsigned)gx < (unsigned)WW)
            v = xb[c * LPIX + gy * WW + gx];
        s_xpad[c][ry][rx] = v;
    }
    for (int i = tid; i < TPX * OUTC * 9; i += blockDim.x) {
        const int px = i / (OUTC * 9);
        const int r  = i % (OUTC * 9);
        s_dot[px][r / 9][r % 9] = 0.f;
        s_sq [px][r / 9][r % 9] = 0.f;
    }
    __syncthreads();

    for (int i = tid; i < CIN * TPX; i += blockDim.x) {
        const int c = i >> 4, px = i & 15;
        s_xcol[c][px] = (_Float16)s_xpad[c][1][px + 1];
    }
    __syncthreads();

    // ---- Per-group B fragments (K = 16 real + 16 zero pad) x 16 pixels
    v16h bG0 = {}, bG1 = {}, bG2 = {}, bG3 = {};
    {
        const int kb = (lane & 16) ? 8 : 0;
#pragma unroll
        for (int h = 0; h < 8; ++h) {
            bG0[h] = s_xcol[ 0 + kb + h][lane & 15];
            bG1[h] = s_xcol[16 + kb + h][lane & 15];
            bG2[h] = s_xcol[32 + kb + h][lane & 15];
            bG3[h] = s_xcol[48 + kb + h][lane & 15];
        }
    }

    const int px_c  = lane & 15;         // column owned by this lane in C tiles
    const int hi    = (lane >> 4) * 8;   // row offset of this lane's C half
    const int dbase = (lane & 16) ? 8 : 0;

    // ---- Phase 1a: pred_proj (64 M-tiles; all in group 3) -> packed f16 LDS
    // unroll 2: two independent WMMA streams hide the WMMA->VALU hazard window
#pragma unroll 2
    for (int it = 0; it < 16; ++it) {
        const int mt = wave + 4 * it;
        const int r0 = WSZ + mt * 16;
        v16h a = wp_afrag(Wp, r0, lane);
        const float4 bq0 = *reinterpret_cast<const float4*>(bp + r0 + hi);
        const float4 bq1 = *reinterpret_cast<const float4*>(bp + r0 + hi + 4);
        v8f c = {};
        c = __builtin_amdgcn_wmma_f32_16x16x32_f16(false, a, false, bG3,
                                                   (short)0, c, false, false);
        v8h hv;
        hv[0] = (_Float16)(c[0] + bq0.x); hv[1] = (_Float16)(c[1] + bq0.y);
        hv[2] = (_Float16)(c[2] + bq0.z); hv[3] = (_Float16)(c[3] + bq0.w);
        hv[4] = (_Float16)(c[4] + bq1.x); hv[5] = (_Float16)(c[5] + bq1.y);
        hv[6] = (_Float16)(c[6] + bq1.z); hv[7] = (_Float16)(c[7] + bq1.w);
        // rows r0..r0+15 are one o (16-aligned block): o = mt, d = hi..hi+7 contiguous
        *reinterpret_cast<v8h*>(s_proj + px_c * PROJ_PXH + mt * 16 + hi) = hv;  // ds_store_b128
    }
    // ---- Phase 1b: dyn_b (4 M-tiles, one per wave) -> padded f32 LDS
    {
        const int r0 = WSZ + PSZ + wave * 16;
        v16h a = wp_afrag(Wp, r0, lane);
        const float4 bq0 = *reinterpret_cast<const float4*>(bp + r0 + hi);
        const float4 bq1 = *reinterpret_cast<const float4*>(bp + r0 + hi + 4);
        v8f c = {};
        c = __builtin_amdgcn_wmma_f32_16x16x32_f16(false, a, false, bG3,
                                                   (short)0, c, false, false);
        float* dst = s_dynb + px_c * DYNB_PXF + wave * 16 + hi;
        *reinterpret_cast<float4*>(dst)     = make_float4(c[0] + bq0.x, c[1] + bq0.y,
                                                          c[2] + bq0.z, c[3] + bq0.w);
        *reinterpret_cast<float4*>(dst + 4) = make_float4(c[4] + bq1.x, c[5] + bq1.y,
                                                          c[6] + bq1.z, c[7] + bq1.w);
    }
    __syncthreads();

    // ---- Phase 3: stream pred_w in 16-wide m-chunks; fuse dyn matmul + norm stats
#pragma unroll 1
    for (int m0 = 0; m0 < CKK; m0 += 16) {
        // 3a: pw chunk, rows r = d*576 + m0 + j (each 16-row tile stays in one group)
#pragma unroll 2
        for (int dk = 0; dk < 4; ++dk) {
            const int d  = wave + 4 * dk;
            const int r0 = d * CKK + m0;
            const int g  = r0 / GRP;
            v16h a = wp_afrag(Wp, r0, lane);
            if (m0 + 16 < CKK)   // prefetch next chunk's row of Wp (L2-resident stream)
                __builtin_prefetch(Wp + (size_t)(r0 + 16 + (lane & 15)) * CPG, 0, 1);
            const float4 bq0 = *reinterpret_cast<const float4*>(bp + r0 + hi);
            const float4 bq1 = *reinterpret_cast<const float4*>(bp + r0 + hi + 4);
            v16h bsel = (g == 0) ? bG0 : (g == 1) ? bG1 : (g == 2) ? bG2 : bG3;
            v8f c = {};
            c = __builtin_amdgcn_wmma_f32_16x16x32_f16(false, a, false, bsel,
                                                       (short)0, c, false, false);
            // transposed store [px][j][d]: 3b reads become one ds_load_b128
            _Float16* pwp = s_pw + px_c * PW_PXH + d;
            pwp[(hi + 0) * PW_JH] = (_Float16)(c[0] + bq0.x);
            pwp[(hi + 1) * PW_JH] = (_Float16)(c[1] + bq0.y);
            pwp[(hi + 2) * PW_JH] = (_Float16)(c[2] + bq0.z);
            pwp[(hi + 3) * PW_JH] = (_Float16)(c[3] + bq0.w);
            pwp[(hi + 4) * PW_JH] = (_Float16)(c[4] + bq1.x);
            pwp[(hi + 5) * PW_JH] = (_Float16)(c[5] + bq1.y);
            pwp[(hi + 6) * PW_JH] = (_Float16)(c[6] + bq1.z);
            pwp[(hi + 7) * PW_JH] = (_Float16)(c[7] + bq1.w);
        }
        __syncthreads();

        // 3b: wave-per-pixel dyn(64x16) = proj(64x16) @ pw(16x16); fused dot/sumsq
        const int m  = m0 + px_c;        // this lane's m-column
        const int cc = m / 9;
        const int kk = m - cc * 9;
        const int ki = kk / 3;
        const int kj = kk % 3;
        const v8h z8 = {};
#pragma unroll 1
        for (int pk = 0; pk < 4; ++pk) {
            const int px = wave * 4 + pk;
            // B fragment: halves 0..7 hold K=d=dbase..dbase+7 for col N=j=lane&15; halves 8..15 pad
            v16h bw = {};
            {
                const _Float16* p = s_pw + px * PW_PXH + px_c * PW_JH + dbase;
                const v8h lo  = *reinterpret_cast<const v8h*>(p);
                bw = __builtin_shufflevector(lo, z8, 0, 1, 2, 3, 4, 5, 6, 7,
                                                      8, 9, 10, 11, 12, 13, 14, 15);
            }
            const float patchv = s_xpad[cc][ki][px + kj];
            // unroll 2: pair independent o-tiles so frag loads fill WMMA hazard slots
#pragma unroll 2
            for (int ot = 0; ot < 4; ++ot) {
                const int o0 = ot * 16;
                const v8h av = *reinterpret_cast<const v8h*>(
                    s_proj + px * PROJ_PXH + (o0 + px_c) * 16 + dbase);   // ds_load_b128
                v16h a = __builtin_shufflevector(av, z8, 0, 1, 2, 3, 4, 5, 6, 7,
                                                         8, 9, 10, 11, 12, 13, 14, 15);
                v8f c = {};
                c = __builtin_amdgcn_wmma_f32_16x16x32_f16(false, a, false, bw,
                                                           (short)0, c, false, false);
#pragma unroll
                for (int v = 0; v < 8; ++v) {
                    const int o = o0 + v + hi;
                    const float dv = c[v];
                    atomicAdd(&s_dot[px][o][kk], dv * patchv);
                    atomicAdd(&s_sq [px][o][kk], dv * dv);
                }
            }
        }
        __syncthreads();
    }

    // ---- Phase 4: out[o] = dyn_b[o] + sum_kk dot[o,kk] * rsqrt(max(sq[o,kk],1e-24))
    for (int i = tid; i < OUTC * TPX; i += blockDim.x) {
        const int o = i >> 4, px = i & 15;
        float acc = s_dynb[px * DYNB_PXF + o];
#pragma unroll
        for (int q = 0; q < 9; ++q) {
            acc += s_dot[px][o][q] * rsqrtf(fmaxf(s_sq[px][o][q], 1e-24f));
        }
        Out[(size_t)bi * OUTC * LPIX + o * LPIX + y * WW + x0 + px] = acc;
    }
}

extern "C" void kernel_launch(void* const* d_in, const int* in_sizes, int n_in,
                              void* d_out, int out_size, void* d_ws, size_t ws_size,
                              hipStream_t stream) {
    (void)in_sizes; (void)n_in; (void)d_ws; (void)ws_size; (void)out_size;
    const float* x  = (const float*)d_in[0];
    const float* Wp = (const float*)d_in[1];
    const float* bp = (const float*)d_in[2];
    float* out = (float*)d_out;

    hipFuncSetAttribute(reinterpret_cast<const void*>(dynconv_fused),
                        hipFuncAttributeMaxDynamicSharedMemorySize, SMEM_BYTES);

    dim3 grid(NB * NTILE);   // 288 workgroups (16-pixel tiles)
    dim3 block(128);         // 4 waves (wave32)
    dynconv_fused<<<grid, block, SMEM_BYTES, stream>>>(x, Wp, bp, out);
}
